// GraphModel_34411277976253
// MI455X (gfx1250) — matrix operand order repaired and checked by reading, not compiled, not run on hardware
//
#include <hip/hip_runtime.h>
#include <cmath>

// ---------------------------------------------------------------------------
// MI455X / gfx1250 EdgeConv GNN.
// One wave32 per node: its 16 incoming edges are the 16 rows (M) of the
// v_wmma_f32_16x16x32_f16 A-matrix. Weights are pre-transposed to [N][K] fp16
// so B fragments are two contiguous global b128 loads per lane.
// tanh uses the CDNA5 hardware V_TANH_F32 (or a branchless exp2/rcp fallback).
// ---------------------------------------------------------------------------

typedef __attribute__((ext_vector_type(16))) _Float16 v16h;
typedef __attribute__((ext_vector_type(8)))  _Float16 v8h;
typedef __attribute__((ext_vector_type(4)))  _Float16 v4h;
typedef __attribute__((ext_vector_type(8)))  float    v8f;
typedef __attribute__((ext_vector_type(4)))  float    v4f;

#define NWAVES 4      // waves (nodes) per block
#define GD     128    // gnn_dim
#define KNB    16     // neighbors per node

// ---- fast tanh: hardware trans op on gfx1250, branchless fallback ----------
__device__ __forceinline__ float fast_tanh(float x) {
#if __has_builtin(__builtin_amdgcn_tanhf)
  return __builtin_amdgcn_tanhf(x);
#else
  // tanh(x) = 1 - 2/(exp(2x)+1); v_exp_f32 is exp2, 2*log2(e) = 2.8853900817779268
  float e = __builtin_amdgcn_exp2f(x * 2.885390082f);
  return 1.0f - 2.0f * __builtin_amdgcn_rcpf(e + 1.0f);
#endif
}

// ---- WMMA fragment loaders -------------------------------------------------
// A (16x32 f16, MxK): lane L -> row M=L%16; half=L/16.
//   VGPR0..3 hold K = 8*half .. 8*half+7 (contiguous), VGPR4..7 K = 16+8*half..
__device__ __forceinline__ v16h load_frag_a(const _Float16* p, int half) {
  union { v16h v; v8h h[2]; } u;
  u.h[0] = *(const v8h*)(p + half * 8);
  u.h[1] = *(const v8h*)(p + 16 + half * 8);
  return u.v;
}
// B (32x16 f16, KxN): lane L -> col N=L%16; half=L/16.
//   VGPR0..3 hold K = 8*half .. 8*half+7, VGPR4..7 K = 16+8*half.. (same as A).
__device__ __forceinline__ v16h load_frag_b(const _Float16* p, int half) {
  union { v16h v; v8h h[2]; } u;
  u.h[0] = *(const v8h*)(p + half * 8);
  u.h[1] = *(const v8h*)(p + 16 + half * 8);
  return u.v;
}

__device__ __forceinline__ v8f wmma_f16(v16h a, v16h b, v8f c) {
  return __builtin_amdgcn_wmma_f32_16x16x32_f16(false, a, false, b,
                                                (short)0, c, false, false);
}

// ---- matmul1: [16 x K] @ [K x 128] + b1, tanh -> mid (fp16 in LDS) ---------
template <int KSTEPS, int WSTRIDE>
__device__ __forceinline__ void mlp1_tanh(const _Float16* arow,
                                          const _Float16* __restrict__ w1t,
                                          const float* __restrict__ b1,
                                          _Float16 (*midw)[136],
                                          int rown, int half) {
  for (int nt = 0; nt < 8; ++nt) {
    v8f c = {0.f, 0.f, 0.f, 0.f, 0.f, 0.f, 0.f, 0.f};
    const _Float16* bcol = w1t + (nt * 16 + rown) * WSTRIDE;
#pragma unroll
    for (int kc = 0; kc < KSTEPS; ++kc) {
      v16h a = load_frag_a(arow + kc * 32, half);
      v16h b = load_frag_b(bcol + kc * 32, half);
      c = wmma_f16(a, b, c);
    }
    float bias = b1[nt * 16 + rown];
#pragma unroll
    for (int r = 0; r < 8; ++r) {   // C layout: M = r + 8*half, N = rown
      midw[r + half * 8][nt * 16 + rown] = (_Float16)fast_tanh(c[r] + bias);
    }
  }
}

// ---- matmul2: [16 x 128] @ [128 x 128] + 16*b2, column-sum over 16 rows ----
__device__ __forceinline__ void mlp2_reduce_store(const _Float16* arow,
                                                  const _Float16* __restrict__ w2t,
                                                  const float* __restrict__ b2,
                                                  float* __restrict__ hout_node,
                                                  int rown, int half) {
  for (int nt = 0; nt < 8; ++nt) {
    v8f c = {0.f, 0.f, 0.f, 0.f, 0.f, 0.f, 0.f, 0.f};
    const _Float16* bcol = w2t + (nt * 16 + rown) * GD;
#pragma unroll
    for (int kc = 0; kc < 4; ++kc) {
      v16h a = load_frag_a(arow + kc * 32, half);
      v16h b = load_frag_b(bcol + kc * 32, half);
      c = wmma_f16(a, b, c);
    }
    // segment_sum over the node's 16 edges == sum over all 16 M rows.
    float s = 0.f;
#pragma unroll
    for (int r = 0; r < 8; ++r) s += c[r];   // this lane's 8 rows
    s += __shfl_xor(s, 16, 32);              // combine half 0 + half 1
    if (half == 0)
      hout_node[nt * 16 + rown] = s + 16.0f * b2[nt * 16 + rown];
  }
}

// ---- generic EdgeConv layer (input dim 128, cat dim 256) -------------------
__global__ __launch_bounds__(NWAVES * 32)
void edgeconv_g(const float* __restrict__ hin, const int* __restrict__ srcIdx,
                const _Float16* __restrict__ w1t,  // [128][256] fp16 (N-major)
                const float* __restrict__ b1,
                const _Float16* __restrict__ w2t,  // [128][128] fp16
                const float* __restrict__ b2,
                float* __restrict__ hout, int nNodes) {
  __shared__ __align__(16) _Float16 cat[NWAVES][16][264];  // 264: bank-pad
  __shared__ __align__(16) _Float16 mid[NWAVES][16][136];
  const int lane = threadIdx.x & 31;
  const int w    = threadIdx.x >> 5;
  const int half = lane >> 4;
  const int rown = lane & 15;
  const int node = blockIdx.x * NWAVES + w;
  if (node >= nNodes) return;   // uniform per wave

  // Gather: cat[e] = [ h_i (128) | h_j - h_i (128) ], fp16. 4 floats per lane.
  v4f xi = *(const v4f*)(hin + (size_t)node * GD + lane * 4);
  const int* se = srcIdx + node * KNB;
  for (int e = 0; e < KNB; ++e) {
    int s = se[e];
    v4f xj = *(const v4f*)(hin + (size_t)s * GD + lane * 4);
    v4h hi, hd;
#pragma unroll
    for (int t = 0; t < 4; ++t) {
      hi[t] = (_Float16)xi[t];
      hd[t] = (_Float16)(xj[t] - xi[t]);
    }
    *(v4h*)&cat[w][e][lane * 4]       = hi;
    *(v4h*)&cat[w][e][128 + lane * 4] = hd;
  }

  mlp1_tanh<8, 256>(&cat[w][rown][0], w1t, b1, mid[w], rown, half);
  mlp2_reduce_store(&mid[w][rown][0], w2t, b2, hout + (size_t)node * GD,
                    rown, half);
}

// ---- layer 0 (input dim 7, cat dim 14 zero-padded to K=32) -----------------
__global__ __launch_bounds__(NWAVES * 32)
void edgeconv0(const float* __restrict__ x, const int* __restrict__ srcIdx,
               const _Float16* __restrict__ w1t,  // [128][32] fp16, K padded
               const float* __restrict__ b1,
               const _Float16* __restrict__ w2t,  // [128][128] fp16
               const float* __restrict__ b2,
               float* __restrict__ hout, int nNodes) {
  __shared__ __align__(16) _Float16 cat[NWAVES][16][40];
  __shared__ __align__(16) _Float16 mid[NWAVES][16][136];
  const int lane = threadIdx.x & 31;
  const int w    = threadIdx.x >> 5;
  const int half = lane >> 4;
  const int rown = lane & 15;
  const int node = blockIdx.x * NWAVES + w;
  if (node >= nNodes) return;

  const int* se = srcIdx + node * KNB;
  for (int e = 0; e < KNB; ++e) {
    int s = se[e];
    float v = 0.f;
    if (lane < 7)
      v = x[(size_t)node * 7 + lane];
    else if (lane < 14)
      v = x[(size_t)s * 7 + (lane - 7)] - x[(size_t)node * 7 + (lane - 7)];
    cat[w][e][lane] = (_Float16)v;             // cols 0..31 (>=14 are zero)
    if (lane < 8) cat[w][e][32 + lane] = (_Float16)0.f;  // pad cols 32..39
  }

  mlp1_tanh<1, 32>(&cat[w][rown][0], w1t, b1, mid[w], rown, half);
  mlp2_reduce_store(&mid[w][rown][0], w2t, b2, hout + (size_t)node * GD,
                    rown, half);
}

// ---- final: z=[x,h] (135) -> LayerNorm -> Linear(135,3) --------------------
__global__ __launch_bounds__(256)
void final_ln(const float* __restrict__ x, const float* __restrict__ h,
              const float* __restrict__ g, const float* __restrict__ bb,
              const float* __restrict__ w_out, const float* __restrict__ b_out,
              float* __restrict__ out, int nNodes) {
  const int lane = threadIdx.x & 31;
  const int node = blockIdx.x * (blockDim.x >> 5) + (threadIdx.x >> 5);
  if (node >= nNodes) return;
  const int D = 135;
  float z[5];
  float sum = 0.f, sq = 0.f;
#pragma unroll
  for (int t = 0; t < 5; ++t) {
    int d = lane + t * 32;
    float v = 0.f;
    if (d < D) v = (d < 7) ? x[(size_t)node * 7 + d]
                           : h[(size_t)node * GD + (d - 7)];
    z[t] = v; sum += v; sq += v * v;
  }
#pragma unroll
  for (int o = 16; o > 0; o >>= 1) {
    sum += __shfl_xor(sum, o, 32);
    sq  += __shfl_xor(sq,  o, 32);
  }
  float mu  = sum * (1.f / 135.f);
  float var = sq * (1.f / 135.f) - mu * mu;
  float rs  = rsqrtf(var + 1e-5f);
  float p0 = 0.f, p1 = 0.f, p2 = 0.f;
#pragma unroll
  for (int t = 0; t < 5; ++t) {
    int d = lane + t * 32;
    if (d < D) {
      float zn = (z[t] - mu) * rs * g[d] + bb[d];
      p0 += zn * w_out[d * 3 + 0];
      p1 += zn * w_out[d * 3 + 1];
      p2 += zn * w_out[d * 3 + 2];
    }
  }
#pragma unroll
  for (int o = 16; o > 0; o >>= 1) {
    p0 += __shfl_xor(p0, o, 32);
    p1 += __shfl_xor(p1, o, 32);
    p2 += __shfl_xor(p2, o, 32);
  }
  if (lane == 0) {
    out[(size_t)node * 3 + 0] = p0 + b_out[0];
    out[(size_t)node * 3 + 1] = p1 + b_out[1];
    out[(size_t)node * 3 + 2] = p2 + b_out[2];
  }
}

// ---- weight prep: fp32 [K x N] -> fp16 [N x Kpad] (zero pad) ---------------
__global__ __launch_bounds__(256)
void transpose_pad(const float* __restrict__ src, _Float16* __restrict__ dst,
                   int Kdim, int Ndim, int Kpad) {
  int i = blockIdx.x * blockDim.x + threadIdx.x;
  if (i >= Ndim * Kpad) return;
  int n = i / Kpad, k = i % Kpad;
  float v = (k < Kdim) ? src[(size_t)k * Ndim + n] : 0.f;
  dst[i] = (_Float16)v;
}

// ---------------------------------------------------------------------------
extern "C" void kernel_launch(void* const* d_in, const int* in_sizes, int n_in,
                              void* d_out, int out_size, void* d_ws,
                              size_t ws_size, hipStream_t stream) {
  const float* x     = (const float*)d_in[0];
  const int*   eidx  = (const int*)d_in[1];     // [2, E]; row0 = src
  const float* w1_0  = (const float*)d_in[2];   // (14,128)
  const float* b1_0  = (const float*)d_in[3];
  const float* w2_0  = (const float*)d_in[4];   // (128,128)
  const float* b2_0  = (const float*)d_in[5];
  const float* w1s   = (const float*)d_in[6];   // (3,256,128)
  const float* b1s   = (const float*)d_in[7];   // (3,128)
  const float* w2s   = (const float*)d_in[8];   // (3,128,128)
  const float* b2s   = (const float*)d_in[9];   // (3,128)
  const float* ln_g  = (const float*)d_in[10];
  const float* ln_b  = (const float*)d_in[11];
  const float* w_out = (const float*)d_in[12];  // (135,3)
  const float* b_out = (const float*)d_in[13];
  float* out = (float*)d_out;

  const int N = in_sizes[0] / 7;
  const int* srcIdx = eidx;  // first E entries of edge_index

  // workspace layout
  size_t off = 0;
  auto take = [&](size_t bytes) -> void* {
    void* p = (char*)d_ws + off;
    off += (bytes + 255) & ~(size_t)255;
    return p;
  };
  _Float16* w1_0t = (_Float16*)take((size_t)128 * 32 * 2);
  _Float16* w2_0t = (_Float16*)take((size_t)128 * 128 * 2);
  _Float16* w1st  = (_Float16*)take((size_t)3 * 128 * 256 * 2);
  _Float16* w2st  = (_Float16*)take((size_t)3 * 128 * 128 * 2);
  float*    hA    = (float*)take((size_t)N * GD * 4);
  float*    hB    = (float*)take((size_t)N * GD * 4);

  auto tp = [&](const float* s, _Float16* d, int K, int Nn, int Kp) {
    int total = Nn * Kp;
    transpose_pad<<<(total + 255) / 256, 256, 0, stream>>>(s, d, K, Nn, Kp);
  };
  tp(w1_0, w1_0t, 14, 128, 32);
  tp(w2_0, w2_0t, 128, 128, 128);
  for (int l = 0; l < 3; ++l) {
    tp(w1s + (size_t)l * 256 * 128, w1st + (size_t)l * 128 * 256, 256, 128, 256);
    tp(w2s + (size_t)l * 128 * 128, w2st + (size_t)l * 128 * 128, 128, 128, 128);
  }

  const int blocks = (N + NWAVES - 1) / NWAVES;
  edgeconv0<<<blocks, NWAVES * 32, 0, stream>>>(x, srcIdx, w1_0t, b1_0,
                                                w2_0t, b2_0, hA, N);
  // 3 more message-passing steps, ping-ponging hA <-> hB
  edgeconv_g<<<blocks, NWAVES * 32, 0, stream>>>(
      hA, srcIdx, w1st + 0 * 128 * 256, b1s + 0 * 128,
      w2st + 0 * 128 * 128, b2s + 0 * 128, hB, N);
  edgeconv_g<<<blocks, NWAVES * 32, 0, stream>>>(
      hB, srcIdx, w1st + 1 * 128 * 256, b1s + 1 * 128,
      w2st + 1 * 128 * 128, b2s + 1 * 128, hA, N);
  edgeconv_g<<<blocks, NWAVES * 32, 0, stream>>>(
      hA, srcIdx, w1st + 2 * 128 * 256, b1s + 2 * 128,
      w2st + 2 * 128 * 128, b2s + 2 * 128, hB, N);

  final_ln<<<(N + 7) / 8, 256, 0, stream>>>(x, hB, ln_g, ln_b, w_out, b_out,
                                            out, N);
}
